// WassersteinSignatureLoss_70772471104022
// MI455X (gfx1250) — matrix operand order repaired and checked by reading, not compiled, not run on hardware
//
#include <hip/hip_runtime.h>
#include <math.h>

// ---------------------------------------------------------------------------
// Sinkhorn divergence (geomloss, p=2) fully fused for gfx1250.
// Each softmin = flash-style streaming logsumexp over the implicit 8192x8192
// cost matrix; dot products via v_wmma_f32_16x16x32_bf16 with a 2-term bf16
// hi/lo split (hi*hi + hi*lo + lo*hi) for near-fp32 accuracy.
// Depth-2 software pipeline: loads(i+1) | wmma(i) | lse-update(i-1), so the
// elementwise VALU co-executes with the XDL WMMA chain and the WMMA->VALU
// RAW hazard is a full iteration away (no v_nop stalls).
// ---------------------------------------------------------------------------

typedef __attribute__((ext_vector_type(16))) __bf16 v16bf;
typedef __attribute__((ext_vector_type(8)))  float  v8f;

#define LOG2E 1.4426950408889634f
#define LN2   0.6931471805599453f

__device__ __forceinline__ unsigned short f2bf(float f) {
  unsigned int u = __float_as_uint(f);
  return (unsigned short)((u + 0x7fffu + ((u >> 16) & 1u)) >> 16);
}
__device__ __forceinline__ float bfval(unsigned short h) {
  return __uint_as_float(((unsigned int)h) << 16);
}

// A fragment (16x32 bf16): lane L holds row M=L%16; lanes 0-15 K={0..7,16..23},
// lanes 16-31 K={8..15,24..31} (ISA 7.12.2). Two 16B loads at +0 / +16 elems.
__device__ __forceinline__ v16bf ldA(const unsigned short* p) {
  union { v16bf v; uint4 u[2]; } t;
  t.u[0] = *reinterpret_cast<const uint4*>(p);
  t.u[1] = *reinterpret_cast<const uint4*>(p + 16);
  return t.v;
}
// B fragment (32x16 bf16): lane L holds col N=L%16; lanes 0-15 K={0..7,16..23},
// lanes 16-31 K={8..15,24..31} (same as A). Two 16B loads at +0 / +16 elems.
__device__ __forceinline__ v16bf ldB(const unsigned short* p) {
  union { v16bf v; uint4 u[2]; } t;
  t.u[0] = *reinterpret_cast<const uint4*>(p);
  t.u[1] = *reinterpret_cast<const uint4*>(p + 16);
  return t.v;
}

// Split fp32 rows into bf16 hi/lo and compute row squared norms.
__global__ void sink_prep_kernel(const float* __restrict__ X,
                                 unsigned short* __restrict__ hi,
                                 unsigned short* __restrict__ lo,
                                 float* __restrict__ sq, int nrows) {
  int row = blockIdx.x * blockDim.x + threadIdx.x;
  if (row >= nrows) return;
  float acc = 0.0f;
#pragma unroll 8
  for (int k = 0; k < 64; ++k) {
    float v = X[row * 64 + k];
    acc = fmaf(v, v, acc);
    unsigned short h = f2bf(v);
    hi[row * 64 + k] = h;
    lo[row * 64 + k] = f2bf(v - bfval(h));
  }
  sq[row] = acc;
}

// Fused per-column bias (log2 domain): tcol[j] = c_h + c_g*pot[j] + c_csq*csq[j]
__global__ void sink_bias_kernel(const float* __restrict__ pot,
                                 const float* __restrict__ csq,
                                 float* __restrict__ tcol,
                                 float c_h, float c_g, float c_csq, int n) {
  int i = blockIdx.x * blockDim.x + threadIdx.x;
  if (i < n) tcol[i] = fmaf(c_g, pot[i], fmaf(c_csq, csq[i], c_h));
}

struct BFrag {
  v16bf bh0, bl0, bh1, bl1;  // hi/lo split, two K=32 chunks
  float t2;                  // fused column bias for this lane's column
};

// One online-LSE step per logit: e = 2^(min-max), branchless merge.
// Raw v_exp_f32 (args <= 0; denorm flush of negligible terms is fine).
__device__ __forceinline__ void lse_update(float m[8], float s[8],
                                           const v8f& d, float t2, float c_dot) {
#pragma unroll
  for (int r = 0; r < 8; ++r) {
    float l  = fmaf(d[r], c_dot, t2);
    float mo = m[r];
    float mn = fmaxf(mo, l);
    float e  = __builtin_amdgcn_exp2f(fminf(mo, l) - mn);
    s[r] = (l > mo) ? fmaf(s[r], e, 1.0f) : (s[r] + e);
    m[r] = mn;
  }
}

// Fused softmin: block = 256 thr = 8 waves; 32 rows/block x 4 col-segments.
__global__ __launch_bounds__(256) void sink_softmin_kernel(
    const unsigned short* __restrict__ Rhi, const unsigned short* __restrict__ Rlo,
    const float* __restrict__ rsq,
    const unsigned short* __restrict__ Chi, const unsigned short* __restrict__ Clo,
    const float* __restrict__ tcol, float* __restrict__ out,
    float c_dot, float eps_ln2, int ncols) {
  __shared__ float2 red[32][64];

  const int tid  = threadIdx.x;
  const int wave = tid >> 5, lane = tid & 31;
  const int half = lane >> 4, lr = lane & 15;
  const int rowGroup = wave >> 2, seg = wave & 3;
  const int rowBase  = blockIdx.x * 32 + rowGroup * 16;

  // A fragments: 16 rows x K=64, hi/lo split, loaded once per wave.
  v16bf ahi[2], alo[2];
  {
    const int arow = rowBase + lr;
    const unsigned short* bh = Rhi + (size_t)arow * 64 + half * 8;
    const unsigned short* bl = Rlo + (size_t)arow * 64 + half * 8;
    ahi[0] = ldA(bh);      alo[0] = ldA(bl);
    ahi[1] = ldA(bh + 32); alo[1] = ldA(bl + 32);
  }

  const int segLen = ncols >> 2;         // columns per segment
  const int nch    = segLen >> 4;        // 16-col chunks per segment
  const int colSeg = seg * segLen;

  auto loadB = [&](int ci) -> BFrag {
    const int bcol = colSeg + ci * 16 + lr;
    const unsigned short* pbh = Chi + (size_t)bcol * 64 + half * 8;
    const unsigned short* pbl = Clo + (size_t)bcol * 64 + half * 8;
    __builtin_prefetch(pbh + 16 * 64, 0, 3);  // next tile
    BFrag f;
    f.bh0 = ldB(pbh);      f.bl0 = ldB(pbl);
    f.bh1 = ldB(pbh + 32); f.bl1 = ldB(pbl + 32);
    f.t2  = tcol[bcol];
    return f;
  };
  auto mm = [&](const BFrag& f) -> v8f {
    v8f d = {};
    d = __builtin_amdgcn_wmma_f32_16x16x32_bf16(false, ahi[0], false, f.bh0, (short)0, d, false, false);
    d = __builtin_amdgcn_wmma_f32_16x16x32_bf16(false, ahi[0], false, f.bl0, (short)0, d, false, false);
    d = __builtin_amdgcn_wmma_f32_16x16x32_bf16(false, alo[0], false, f.bh0, (short)0, d, false, false);
    d = __builtin_amdgcn_wmma_f32_16x16x32_bf16(false, ahi[1], false, f.bh1, (short)0, d, false, false);
    d = __builtin_amdgcn_wmma_f32_16x16x32_bf16(false, ahi[1], false, f.bl1, (short)0, d, false, false);
    d = __builtin_amdgcn_wmma_f32_16x16x32_bf16(false, alo[1], false, f.bh1, (short)0, d, false, false);
    return d;
  };

  float m[8], s[8];
#pragma unroll
  for (int r = 0; r < 8; ++r) { m[r] = -3.0e38f; s[r] = 0.0f; }

  // ---- depth-2 software pipeline over nch chunks ----
  BFrag fc = loadB(0);
  BFrag fn = loadB(1);
  v8f   dp = mm(fc);          // chunk 0 logits in flight
  float t2p = fc.t2;
  fc = fn;
  for (int ci = 2; ci < nch; ++ci) {
    fn = loadB(ci);                      // loads for chunk ci (next)
    v8f   d   = mm(fc);                  // WMMA chain for chunk ci-1
    float t2d = fc.t2;
    lse_update(m, s, dp, t2p, c_dot);    // elementwise for chunk ci-2 (co-exec)
    dp = d; t2p = t2d;
    fc = fn;
  }
  {
    v8f   d   = mm(fc);                  // chunk nch-1
    float t2d = fc.t2;
    lse_update(m, s, dp, t2p, c_dot);
    lse_update(m, s, d, t2d, c_dot);
  }

  // D layout: lane holds col N=lr; VGPR r -> row r + 8*half (ISA 7.12.2).
#pragma unroll
  for (int r = 0; r < 8; ++r)
    red[rowGroup * 16 + half * 8 + r][seg * 16 + lr] = make_float2(m[r], s[r]);
  __syncthreads();

  if (tid < 32) {
    float M = -3.0e38f;
#pragma unroll 4
    for (int k = 0; k < 64; ++k) M = fmaxf(M, red[tid][k].x);
    float S = 0.0f;
#pragma unroll 4
    for (int k = 0; k < 64; ++k)
      S += red[tid][k].y * __builtin_amdgcn_exp2f(red[tid][k].x - M);
    const int row = blockIdx.x * 32 + tid;
    out[row] = 0.5f * rsq[row] - eps_ln2 * (M + log2f(S));
  }
}

// a = 0.5*(a + b)
__global__ void sink_avg_kernel(float* __restrict__ a, const float* __restrict__ b, int n) {
  int i = blockIdx.x * blockDim.x + threadIdx.x;
  if (i < n) a[i] = 0.5f * (a[i] + b[i]);
}

// out[0] = mean(fba - faa) + mean(gab - gbb)
__global__ void sink_final_kernel(const float* __restrict__ fba, const float* __restrict__ faa,
                                  const float* __restrict__ gab, const float* __restrict__ gbb,
                                  float* __restrict__ out, int n, int mcols) {
  __shared__ float sh1[256], sh2[256];
  float a1 = 0.0f, a2 = 0.0f;
  for (int i = threadIdx.x; i < n; i += 256) a1 += fba[i] - faa[i];
  for (int i = threadIdx.x; i < mcols; i += 256) a2 += gab[i] - gbb[i];
  sh1[threadIdx.x] = a1; sh2[threadIdx.x] = a2;
  __syncthreads();
  for (int st = 128; st > 0; st >>= 1) {
    if (threadIdx.x < st) {
      sh1[threadIdx.x] += sh1[threadIdx.x + st];
      sh2[threadIdx.x] += sh2[threadIdx.x + st];
    }
    __syncthreads();
  }
  if (threadIdx.x == 0) out[0] = sh1[0] / (float)n + sh2[0] / (float)mcols;
}

extern "C" void kernel_launch(void* const* d_in, const int* in_sizes, int n_in,
                              void* d_out, int out_size, void* d_ws, size_t ws_size,
                              hipStream_t stream) {
  (void)n_in; (void)out_size; (void)ws_size;
  const float* x = (const float*)d_in[0];
  const float* y = (const float*)d_in[1];
  float* out = (float*)d_out;

  const int D = 64;
  const int N = in_sizes[0] / D;   // 8192
  const int M = in_sizes[1] / D;   // 8192

  // workspace carve-out (256B aligned)
  char* ws = (char*)d_ws;
  size_t off = 0;
  auto give = [&](size_t bytes) -> char* {
    char* p = ws + off;
    off += (bytes + 255) & ~(size_t)255;
    return p;
  };
  unsigned short* xhi = (unsigned short*)give((size_t)N * D * 2);
  unsigned short* xlo = (unsigned short*)give((size_t)N * D * 2);
  unsigned short* yhi = (unsigned short*)give((size_t)M * D * 2);
  unsigned short* ylo = (unsigned short*)give((size_t)M * D * 2);
  float* x2   = (float*)give((size_t)N * 4);
  float* y2   = (float*)give((size_t)M * 4);
  float* f_aa = (float*)give((size_t)N * 4);
  float* g_bb = (float*)give((size_t)M * 4);
  float* g_ab = (float*)give((size_t)M * 4);
  float* f_ba = (float*)give((size_t)N * 4);
  float* t_ft = (float*)give((size_t)N * 4);
  float* t_gt = (float*)give((size_t)M * 4);
  float* t_fs = (float*)give((size_t)N * 4);
  float* t_gs = (float*)give((size_t)M * 4);
  float* tcol = (float*)give((size_t)(N > M ? N : M) * 4);

  sink_prep_kernel<<<(N + 255) / 256, 256, 0, stream>>>(x, xhi, xlo, x2, N);
  sink_prep_kernel<<<(M + 255) / 256, 256, 0, stream>>>(y, yhi, ylo, y2, M);

  // geomloss eps schedule: [1024,256,64,16,4,1,.25,.0625,.015625,.00390625,.0025]
  double epsl[64]; int ne = 0;
  epsl[ne++] = 32.0 * 32.0;
  double e = 32.0;
  while (e > 0.05) { e = (e * 0.5 > 0.05) ? e * 0.5 : 0.05; epsl[ne++] = e * e; }

  const float alog = -logf((float)N);  // N == M -> a_log == b_log

  // softmin(eps, C_{rows,cols}, hscalar + use_pot*pot/eps) -> o
  auto softmin = [&](double eps,
                     const unsigned short* Rhi, const unsigned short* Rlo,
                     const float* rsq, int nrows,
                     const unsigned short* Chi_, const unsigned short* Clo_,
                     const float* csq, int ncols,
                     const float* pot, int use_pot, float* o) {
    float inv_eps = (float)(1.0 / eps);
    sink_bias_kernel<<<(ncols + 255) / 256, 256, 0, stream>>>(
        pot ? pot : csq, csq, tcol,
        alog * LOG2E,                        // c_h
        use_pot ? inv_eps * LOG2E : 0.0f,    // c_g
        -0.5f * inv_eps * LOG2E,             // c_csq
        ncols);
    sink_softmin_kernel<<<nrows / 32, 256, 0, stream>>>(
        Rhi, Rlo, rsq, Chi_, Clo_, tcol, o,
        inv_eps * LOG2E,                     // c_dot
        (float)(eps * (double)LN2),          // eps*ln2
        ncols);
  };

  // ---- init at eps0 ----
  softmin(epsl[0], xhi, xlo, x2, N, xhi, xlo, x2, N, nullptr, 0, f_aa);  // C_xx, a_log
  softmin(epsl[0], yhi, ylo, y2, M, yhi, ylo, y2, M, nullptr, 0, g_bb);  // C_yy, b_log
  softmin(epsl[0], yhi, ylo, y2, M, xhi, xlo, x2, N, nullptr, 0, g_ab);  // C_yx, a_log
  softmin(epsl[0], xhi, xlo, x2, N, yhi, ylo, y2, M, nullptr, 0, f_ba);  // C_xy, b_log

  // ---- eps-scaling loop with symmetric averaged updates ----
  for (int i = 0; i < ne; ++i) {
    double eps = epsl[i];
    softmin(eps, xhi, xlo, x2, N, yhi, ylo, y2, M, g_ab, 1, t_ft);  // ft
    softmin(eps, yhi, ylo, y2, M, xhi, xlo, x2, N, f_ba, 1, t_gt);  // gt
    softmin(eps, xhi, xlo, x2, N, xhi, xlo, x2, N, f_aa, 1, t_fs);  // fs
    softmin(eps, yhi, ylo, y2, M, yhi, ylo, y2, M, g_bb, 1, t_gs);  // gs
    sink_avg_kernel<<<(N + 255) / 256, 256, 0, stream>>>(f_ba, t_ft, N);
    sink_avg_kernel<<<(M + 255) / 256, 256, 0, stream>>>(g_ab, t_gt, M);
    sink_avg_kernel<<<(N + 255) / 256, 256, 0, stream>>>(f_aa, t_fs, N);
    sink_avg_kernel<<<(M + 255) / 256, 256, 0, stream>>>(g_bb, t_gs, M);
  }

  // ---- final non-averaged extrapolation at eps = blur^p ----
  double epsf = epsl[ne - 1];
  softmin(epsf, xhi, xlo, x2, N, yhi, ylo, y2, M, g_ab, 1, t_ft);  // f_ba_f
  softmin(epsf, yhi, ylo, y2, M, xhi, xlo, x2, N, f_ba, 1, t_gt);  // g_ab_f
  softmin(epsf, xhi, xlo, x2, N, xhi, xlo, x2, N, f_aa, 1, t_fs);  // f_aa_f
  softmin(epsf, yhi, ylo, y2, M, yhi, ylo, y2, M, g_bb, 1, t_gs);  // g_bb_f

  sink_final_kernel<<<1, 256, 0, stream>>>(t_ft, t_fs, t_gt, t_gs, out, N, M);
}
